// ProtoNet_41317585387554
// MI455X (gfx1250) — matrix operand order repaired and checked by reading, not compiled, not run on hardware
//
#include <hip/hip_runtime.h>
#include <hip/hip_bf16.h>
#include <stdint.h>

// ---------------------------------------------------------------------------
// Problem constants (match setup_inputs: N=10, K=5, total_Q=150)
// ---------------------------------------------------------------------------
#define HIDDEN    768
#define MAX_LEN   128
#define CONV_NUM  8
#define CONV_SIZE 96
#define TPOS      673          // HIDDEN - CONV_SIZE + 1
#define NTILES    43           // ceil(688/16); 688 = TPOS padded to 16
#define FC_IN     5384         // TPOS * CONV_NUM
#define KPAD      5408         // FC_IN padded to multiple of 32
#define KSTEPS    169          // KPAD / 32
#define BS        400
#define BQ        1200
#define BTOT      1600
#define NWAY      10
#define KSHOT     5
#define TOTQ      150
#define BE        8            // BS / (NWAY*KSHOT)

typedef __attribute__((ext_vector_type(16))) __bf16 v16bf;
typedef __attribute__((ext_vector_type(8)))  float  v8f;

union BF16x16 {
    v16bf v;
    unsigned short u[16];
};

__device__ __forceinline__ unsigned short f32_to_bf16(float f) {
    union { float f; unsigned u; } x; x.f = f;
    unsigned r = x.u + 0x7FFFu + ((x.u >> 16) & 1u);   // round-to-nearest-even
    return (unsigned short)(r >> 16);
}

// ---------------------------------------------------------------------------
// Kernel: float -> bf16 convert (flat)
// ---------------------------------------------------------------------------
__global__ void cvt_bf16_kernel(const float* __restrict__ in,
                                unsigned short* __restrict__ out, int n) {
    int i = blockIdx.x * blockDim.x + threadIdx.x;
    if (i < n) out[i] = f32_to_bf16(in[i]);
}

// float -> bf16 with row padding (fc_W: [rows][incols] -> [rows][outcols], pad=0)
__global__ void cvt_bf16_pad_kernel(const float* __restrict__ in,
                                    unsigned short* __restrict__ out,
                                    int rows, int incols, int outcols) {
    int i = blockIdx.x * blockDim.x + threadIdx.x;
    int total = rows * outcols;
    if (i >= total) return;
    int r = i / outcols, c = i - r * outcols;
    out[i] = (c < incols) ? f32_to_bf16(in[(size_t)r * incols + c]) : (unsigned short)0;
}

// ---------------------------------------------------------------------------
// Kernel: ragged entity mean  (mean of x[b, s:max(e,s+1), :]) -> bf16 [B][768]
// ---------------------------------------------------------------------------
__global__ void entity_mean_kernel(const float* __restrict__ x,
                                   const int* __restrict__ s,
                                   const int* __restrict__ e,
                                   unsigned short* __restrict__ out) {
    int b = blockIdx.x;
    int s0 = s[b];
    int e0 = e[b];
    if (e0 < s0 + 1) e0 = s0 + 1;
    float inv = 1.0f / (float)(e0 - s0);
    const float* xb = x + (size_t)b * MAX_LEN * HIDDEN;
    for (int h = threadIdx.x; h < HIDDEN; h += blockDim.x) {
        float acc = 0.0f;
        for (int l = s0; l < e0; ++l) acc += xb[(size_t)l * HIDDEN + h];
        out[(size_t)b * HIDDEN + h] = f32_to_bf16(acc * inv);
    }
}

// ---------------------------------------------------------------------------
// Kernel: per-sample grouped conv (head+tail stacked as 16xK A-matrix) via
// v_wmma_f32_16x16x32_bf16, then ReLU + concat-avgpool(2) -> flat[b][KPAD] bf16
// One workgroup (4 waves) per sample; x and weights staged in LDS.
// Each wave runs TWO independent accumulator chains (tile pair nt, nt+4) with
// interleaved WMMAs so the D->C RAW hazard is covered by co-executed WMMAs
// instead of v_nop bubbles. Wave index is readfirstlane'd (scalar loop,
// EXEC==all-ones at every WMMA).
// ---------------------------------------------------------------------------
__global__ __launch_bounds__(128) void conv_pool_kernel(
    const unsigned short* __restrict__ emb,    // [B][768] bf16
    const unsigned short* __restrict__ headw,  // [B][768] bf16
    const unsigned short* __restrict__ tailw,  // [B][768] bf16
    unsigned short* __restrict__ flat,         // [BTOT][KPAD] bf16
    int rowoff) {
    __shared__ unsigned short xs[800];             // 768 + zero pad
    __shared__ unsigned short ws[16 * CONV_SIZE];  // rows 0..7 head, 8..15 tail
    __shared__ float conv[16][688];                // relu(conv) staging

    int b = blockIdx.x;
    const unsigned short* xg = emb + (size_t)b * HIDDEN;
    for (int i = threadIdx.x; i < 800; i += blockDim.x)
        xs[i] = (i < HIDDEN) ? xg[i] : (unsigned short)0;
    for (int i = threadIdx.x; i < 16 * CONV_SIZE; i += blockDim.x) {
        int r = i / CONV_SIZE, k = i - r * CONV_SIZE;
        ws[i] = (r < 8) ? headw[(size_t)b * HIDDEN + r * CONV_SIZE + k]
                        : tailw[(size_t)b * HIDDEN + (r - 8) * CONV_SIZE + k];
    }
    __syncthreads();

    int wave = __builtin_amdgcn_readfirstlane((int)(threadIdx.x >> 5));  // SGPR
    int lane = threadIdx.x & 31;
    int half = lane >> 4;        // 0: lanes 0-15, 1: lanes 16-31
    int lm   = lane & 15;

    // A fragments (16x32 bf16, M=16 weight rows), hoisted across N tiles.
    // Layout: lanes 0-15 hold K=0..15, lanes 16-31 K=16..31 (matches B).
    BF16x16 afrag[3];
#pragma unroll
    for (int ks = 0; ks < 3; ++ks) {
        int k0 = ks * 32 + (half ? 16 : 0);
        int k1 = ks * 32 + (half ? 24 : 8);
#pragma unroll
        for (int i = 0; i < 8; ++i) {
            afrag[ks].u[i]     = ws[lm * CONV_SIZE + k0 + i];
            afrag[ks].u[8 + i] = ws[lm * CONV_SIZE + k1 + i];
        }
    }

    // Tile pairs (nt, nt+4); residues mod 8 across 4 waves cover 0..42 once.
    for (int nt = wave; nt < NTILES; nt += 8) {
        int ntB  = nt + 4;
        int hasB = (ntB < NTILES);            // wave-uniform scalar
        int t0 = nt * 16 + lm;
        int t1 = hasB ? (ntB * 16 + lm) : t0; // dummy chain aims at t0
        v8f acc0 = {}, acc1 = {};
#pragma unroll
        for (int ks = 0; ks < 3; ++ks) {
            // B (32x16): B[k][t] = x[t + k]; lanes 0-15 K=0..15, lanes 16-31 K=16..31
            int kb = ks * 32 + (half ? 16 : 0);
            BF16x16 b0, b1;
#pragma unroll
            for (int i = 0; i < 16; ++i) {
                b0.u[i] = xs[t0 + kb + i];
                b1.u[i] = xs[t1 + kb + i];
            }
            acc0 = __builtin_amdgcn_wmma_f32_16x16x32_bf16(
                false, afrag[ks].v, false, b0.v, (short)0, acc0, false, false);
            acc1 = __builtin_amdgcn_wmma_f32_16x16x32_bf16(
                false, afrag[ks].v, false, b1.v, (short)0, acc1, false, false);
        }
        // C/D layout: element v of lane L -> row v + 8*(L>=16), col L%16
#pragma unroll
        for (int v = 0; v < 8; ++v)
            conv[v + half * 8][t0] = fmaxf(acc0[v], 0.0f);
        if (hasB) {
#pragma unroll
            for (int v = 0; v < 8; ++v)
                conv[v + half * 8][t1] = fmaxf(acc1[v], 0.0f);
        }
    }
    __syncthreads();

    // concat([h,t], time) then avgpool(k=2,s=2): straddles boundary at index 673
    unsigned short* frow = flat + (size_t)(rowoff + b) * KPAD;
    for (int i = threadIdx.x; i < KPAD; i += blockDim.x) {
        unsigned short r = 0;
        if (i < FC_IN) {
            int c = i / TPOS, j = i - c * TPOS;
            int i0 = 2 * j, i1 = 2 * j + 1;
            float v0 = (i0 < TPOS) ? conv[c][i0] : conv[8 + c][i0 - TPOS];
            float v1 = (i1 < TPOS) ? conv[c][i1] : conv[8 + c][i1 - TPOS];
            r = f32_to_bf16(0.5f * (v0 + v1));
        }
        frow[i] = r;
    }
}

// ---------------------------------------------------------------------------
// Kernel: FC GEMM  out[1600][768] = flat[1600][KPAD] @ fcW[768][KPAD]^T + b
// Register-blocked 1x4: each wave holds ONE A fragment and FOUR accumulators,
// streaming 4 B tiles per K-step (4x A reuse; 4 wmma per A load; independent
// accumulator chains hide the WMMA RAW hazard).
// Block = 4 waves (same M tile); grid (3, 100) covers 48 N tiles x 100 M tiles.
// K is contiguous for both operands -> aligned b128 loads + global_prefetch_b8.
// ---------------------------------------------------------------------------
__global__ __launch_bounds__(128) void fc_gemm_kernel(
    const unsigned short* __restrict__ A,   // flat  [BTOT][KPAD] bf16
    const unsigned short* __restrict__ Bm,  // fc_W  [768][KPAD]  bf16
    const float* __restrict__ bias,         // [768]
    float* __restrict__ out) {              // [BTOT][768]
    int mt   = blockIdx.y;                  // 0..99
    int wave = __builtin_amdgcn_readfirstlane((int)(threadIdx.x >> 5));
    int lane = threadIdx.x & 31;
    int nt0  = (blockIdx.x * 4 + wave) * 4; // first of 4 N tiles, 0..44
    int half = lane >> 4;
    int lm   = lane & 15;

    const unsigned short* arow = A + (size_t)(mt * 16 + lm) * KPAD;
    const unsigned short* brow[4];
#pragma unroll
    for (int j = 0; j < 4; ++j)
        brow[j] = Bm + (size_t)((nt0 + j) * 16 + lm) * KPAD;

    v8f acc0 = {}, acc1 = {}, acc2 = {}, acc3 = {};
    for (int ks = 0; ks < KSTEPS; ++ks) {
        int kb = ks * 32;
        __builtin_prefetch((const void*)(arow + kb + 256), 0, 0);
        // A fragment: lanes 0-15 K=0..15, lanes 16-31 K=16..31
        BF16x16 a;
        *(uint4*)&a.u[0] = *(const uint4*)(arow + kb + (half ? 16 : 0));
        *(uint4*)&a.u[8] = *(const uint4*)(arow + kb + (half ? 24 : 8));
        // 4 B fragments against the same A fragment
        BF16x16 b0, b1, b2, b3;
        {
            const uint4* p = (const uint4*)(brow[0] + kb + (half ? 16 : 0));
            *(uint4*)&b0.u[0] = p[0]; *(uint4*)&b0.u[8] = p[1];
        }
        {
            const uint4* p = (const uint4*)(brow[1] + kb + (half ? 16 : 0));
            *(uint4*)&b1.u[0] = p[0]; *(uint4*)&b1.u[8] = p[1];
        }
        {
            const uint4* p = (const uint4*)(brow[2] + kb + (half ? 16 : 0));
            *(uint4*)&b2.u[0] = p[0]; *(uint4*)&b2.u[8] = p[1];
        }
        {
            const uint4* p = (const uint4*)(brow[3] + kb + (half ? 16 : 0));
            *(uint4*)&b3.u[0] = p[0]; *(uint4*)&b3.u[8] = p[1];
        }
        __builtin_prefetch((const void*)(brow[0] + kb + 256), 0, 0);
        __builtin_prefetch((const void*)(brow[2] + kb + 256), 0, 0);
        acc0 = __builtin_amdgcn_wmma_f32_16x16x32_bf16(
            false, a.v, false, b0.v, (short)0, acc0, false, false);
        acc1 = __builtin_amdgcn_wmma_f32_16x16x32_bf16(
            false, a.v, false, b1.v, (short)0, acc1, false, false);
        acc2 = __builtin_amdgcn_wmma_f32_16x16x32_bf16(
            false, a.v, false, b2.v, (short)0, acc2, false, false);
        acc3 = __builtin_amdgcn_wmma_f32_16x16x32_bf16(
            false, a.v, false, b3.v, (short)0, acc3, false, false);
    }

#pragma unroll
    for (int j = 0; j < 4; ++j) {
        const v8f& acc = (j == 0) ? acc0 : (j == 1) ? acc1 : (j == 2) ? acc2 : acc3;
        int ncol = (nt0 + j) * 16 + lm;
        float bv = bias[ncol];
#pragma unroll
        for (int v = 0; v < 8; ++v) {
            int mrow = mt * 16 + v + half * 8;
            out[(size_t)mrow * HIDDEN + ncol] = acc[v] + bv;
        }
    }
}

// ---------------------------------------------------------------------------
// Kernel: prototype mean over K shots: proto[80][768]
// ---------------------------------------------------------------------------
__global__ void proto_kernel(const float* __restrict__ feat,   // [BTOT][768], rows 0..399 = support
                             float* __restrict__ proto) {      // [BE*NWAY][768]
    int p = blockIdx.x;                                        // e*NWAY + n, 0..79
    for (int h = threadIdx.x; h < HIDDEN; h += blockDim.x) {
        float a = 0.0f;
#pragma unroll
        for (int k = 0; k < KSHOT; ++k)
            a += feat[(size_t)(p * KSHOT + k) * HIDDEN + h];
        proto[(size_t)p * HIDDEN + h] = a * (1.0f / KSHOT);
    }
}

// ---------------------------------------------------------------------------
// Kernel: logits = -||proto - query||^2, append min-1, argmax.
// One block per (episode, query); wave n handles prototype n (wave32 shuffles).
// d_out: logits [8*150*11] f32 then pred [1200] stored as f32.
// ---------------------------------------------------------------------------
__global__ __launch_bounds__(320) void logits_kernel(
    const float* __restrict__ feat,    // [BTOT][768]
    const float* __restrict__ proto,   // [80][768]
    float* __restrict__ outp) {
    int eq = blockIdx.x;               // 0..1199
    int e  = eq / TOTQ;
    int wave = threadIdx.x >> 5;
    int lane = threadIdx.x & 31;
    __shared__ float dist[NWAY];

    const float* q = feat + (size_t)(BS + eq) * HIDDEN;
    const float* p = proto + (size_t)(e * NWAY + wave) * HIDDEN;
    float acc = 0.0f;
    for (int h = lane; h < HIDDEN; h += 32) {
        float d = p[h] - q[h];
        acc += d * d;
    }
#pragma unroll
    for (int off = 16; off; off >>= 1) acc += __shfl_xor(acc, off, 32);
    if (lane == 0) dist[wave] = acc;
    __syncthreads();

    if (threadIdx.x == 0) {
        float l[NWAY];
        float mn = 3.4e38f, best = -3.4e38f;
        int arg = 0;
#pragma unroll
        for (int n = 0; n < NWAY; ++n) {
            l[n] = -dist[n];
            mn = fminf(mn, l[n]);
            if (l[n] > best) { best = l[n]; arg = n; }
        }
        float* row = outp + (size_t)eq * (NWAY + 1);
#pragma unroll
        for (int n = 0; n < NWAY; ++n) row[n] = l[n];
        row[NWAY] = mn - 1.0f;
        outp[(size_t)BE * TOTQ * (NWAY + 1) + eq] = (float)arg;
    }
}

// ---------------------------------------------------------------------------
// Host launcher
// ---------------------------------------------------------------------------
extern "C" void kernel_launch(void* const* d_in, const int* in_sizes, int n_in,
                              void* d_out, int out_size, void* d_ws, size_t ws_size,
                              hipStream_t stream) {
    const float* all_support = (const float*)d_in[0];
    const float* support_emb = (const float*)d_in[1];
    const float* all_query   = (const float*)d_in[2];
    const float* query_emb   = (const float*)d_in[3];
    const float* fc_W        = (const float*)d_in[4];
    const float* fc_b        = (const float*)d_in[5];
    const int* s_eh     = (const int*)d_in[6];
    const int* s_eh_end = (const int*)d_in[7];
    const int* s_et     = (const int*)d_in[8];
    const int* s_et_end = (const int*)d_in[9];
    const int* q_eh     = (const int*)d_in[10];
    const int* q_eh_end = (const int*)d_in[11];
    const int* q_et     = (const int*)d_in[12];
    const int* q_et_end = (const int*)d_in[13];
    (void)in_sizes; (void)n_in; (void)out_size; (void)ws_size;

    // Bump allocator on workspace (256B aligned blocks)
    char* w = (char*)d_ws;
    auto alloc = [&](size_t bytes) -> void* {
        void* p = (void*)w;
        w += (bytes + 255) & ~(size_t)255;
        return p;
    };
    unsigned short* semb  = (unsigned short*)alloc((size_t)BS * HIDDEN * 2);
    unsigned short* qemb  = (unsigned short*)alloc((size_t)BQ * HIDDEN * 2);
    unsigned short* sh    = (unsigned short*)alloc((size_t)BS * HIDDEN * 2);
    unsigned short* st    = (unsigned short*)alloc((size_t)BS * HIDDEN * 2);
    unsigned short* qh    = (unsigned short*)alloc((size_t)BQ * HIDDEN * 2);
    unsigned short* qt    = (unsigned short*)alloc((size_t)BQ * HIDDEN * 2);
    unsigned short* fcWb  = (unsigned short*)alloc((size_t)HIDDEN * KPAD * 2);
    unsigned short* flat  = (unsigned short*)alloc((size_t)BTOT * KPAD * 2);
    float*          feat  = (float*)alloc((size_t)BTOT * HIDDEN * 4);
    float*          proto = (float*)alloc((size_t)BE * NWAY * HIDDEN * 4);

    // 1) precision conversion (bf16) for embeddings + fc weights
    {
        int n = BS * HIDDEN;
        cvt_bf16_kernel<<<(n + 255) / 256, 256, 0, stream>>>(support_emb, semb, n);
        n = BQ * HIDDEN;
        cvt_bf16_kernel<<<(n + 255) / 256, 256, 0, stream>>>(query_emb, qemb, n);
        n = HIDDEN * KPAD;
        cvt_bf16_pad_kernel<<<(n + 255) / 256, 256, 0, stream>>>(fc_W, fcWb, HIDDEN, FC_IN, KPAD);
    }

    // 2) ragged entity means (ragged streaming, ~100 MB total)
    entity_mean_kernel<<<BS, 256, 0, stream>>>(all_support, s_eh, s_eh_end, sh);
    entity_mean_kernel<<<BS, 256, 0, stream>>>(all_support, s_et, s_et_end, st);
    entity_mean_kernel<<<BQ, 256, 0, stream>>>(all_query,   q_eh, q_eh_end, qh);
    entity_mean_kernel<<<BQ, 256, 0, stream>>>(all_query,   q_et, q_et_end, qt);

    // 3) per-sample grouped conv (WMMA, dual accumulator chains) -> flat
    conv_pool_kernel<<<BS, 128, 0, stream>>>(semb, sh, st, flat, 0);
    conv_pool_kernel<<<BQ, 128, 0, stream>>>(qemb, qh, qt, flat, BS);

    // 4) FC GEMM (WMMA, 1x4 register blocking): [1600,5408] x [5408,768]
    fc_gemm_kernel<<<dim3(3, BTOT / 16), 128, 0, stream>>>(flat, fcWb, fc_b, feat);

    // 5) prototypes + logits + argmax
    proto_kernel<<<BE * NWAY, 256, 0, stream>>>(feat, proto);
    logits_kernel<<<BE * TOTQ, 320, 0, stream>>>(feat, proto, (float*)d_out);
}
